// HybridQLSTM_65481071407934
// MI455X (gfx1250) — matrix operand order repaired and checked by reading, not compiled, not run on hardware
//
#include <hip/hip_runtime.h>

#define T_STEPS 512
#define BATCH   128
#define DIM     1024
#define HID     1024
#define KTOT    2048   // D + H
// packed weights: [ntg(256)][kt(64)][lane(32)][j(16)]  bf16  -> 16 MB
#define PACK_ELEMS (256 * 64 * 32 * 16)

typedef __attribute__((ext_vector_type(16))) __bf16 v16bf;
typedef __attribute__((ext_vector_type(8)))  __bf16 v8bf;
typedef __attribute__((ext_vector_type(4)))  __bf16 v4bf;
typedef __attribute__((ext_vector_type(8)))  float  v8f;

// ---------------------------------------------------------------------------
// Pack Wf|Wi|Wg|Wo (fp32, [H, D+H]) into bf16 WMMA-B lane layout.
// B[k][n] = W[n][k].  Lane l: col = l&15, K-range = kt*32 + 16*(l>=16) .. +15.
// ---------------------------------------------------------------------------
__global__ __launch_bounds__(256) void pack_weights(
    const float* __restrict__ Wf, const float* __restrict__ Wi,
    const float* __restrict__ Wg, const float* __restrict__ Wo,
    __bf16* __restrict__ wpack)
{
    int idx = blockIdx.x * 256 + threadIdx.x;
    if (idx >= PACK_ELEMS) return;
    int j    = idx & 15;
    int lane = (idx >> 4) & 31;
    int kt   = (idx >> 9) & 63;
    int ntg  = idx >> 15;                       // 0..255 (gate*64 + n0)
    int n_global = ntg * 16 + (lane & 15);      // 0..4095
    int k        = kt * 32 + ((lane >> 4) << 4) + j;
    int gate = n_global >> 10;
    int row  = n_global & 1023;
    const float* W = (gate == 0) ? Wf : (gate == 1) ? Wi : (gate == 2) ? Wg : Wo;
    wpack[idx] = (__bf16)W[(size_t)row * KTOT + k];
}

__device__ __forceinline__ float sigmoid_f(float x) {
    return 1.0f / (1.0f + __expf(-x));
}
__device__ __forceinline__ float tanh_f(float x) {
    float e = __expf(2.0f * x);
    return (e - 1.0f) / (e + 1.0f);
}

// ---------------------------------------------------------------------------
// One LSTM timestep.  grid = (64 hidden-tiles, 4 batch-groups), block = 64.
// Each wave: 16 batch rows x 16 hidden units, 4 gate accumulators.
// Double-buffered LDS A-staging: stage chunk c+1 while WMMA'ing chunk c.
// ---------------------------------------------------------------------------
__global__ __launch_bounds__(64) void lstm_step(
    const float* __restrict__ x_t,      // [B, D]
    const float* __restrict__ h_prev,   // [B, H]  (unused when first)
    const __bf16* __restrict__ wpack,
    const float* __restrict__ bF_, const float* __restrict__ bI_,
    const float* __restrict__ bG_, const float* __restrict__ bO_,
    float* __restrict__ c_state,        // [B, H] fp32
    float* __restrict__ h_out,          // [B, H] fp32
    int first)
{
    // 2 x (32 rows x 64 k) bf16, row stride 72 (144 B = 36 banks, conflict-free)
    __shared__ __attribute__((aligned(16))) __bf16 smem[2][32 * 72];

    const int tid  = threadIdx.x;
    const int lane = tid & 31;
    const int wave = tid >> 5;                 // 0..1
    const int n0   = blockIdx.x;               // hidden tile 0..63
    const int mg   = blockIdx.y;               // batch group 0..3
    const int mb   = mg * 32 + wave * 16;      // this wave's batch base

    const int srow  = tid >> 1;                // staging row 0..31
    const int shalf = tid & 1;                 // which 32-float half of chunk
    const int sb    = mg * 32 + srow;          // staged global batch row

    const int hi = lane >> 4;
    const int r  = lane & 15;

    v8f accf = {0,0,0,0,0,0,0,0};
    v8f acci = {0,0,0,0,0,0,0,0};
    v8f accg = {0,0,0,0,0,0,0,0};
    v8f acco = {0,0,0,0,0,0,0,0};

    const size_t gate_stride = (size_t)64 * 64 * 32 * 16;  // ntg stride per gate
    const int NCHUNK = KTOT / 64;                          // 32

    // stage chunk c (64 K values, 32 rows) into smem[buf]
    auto stage = [&](int c, int buf) {
        int kg = c * 64 + shalf * 32;
        __bf16* dst = &smem[buf][srow * 72 + shalf * 32];
        bool is_x = (kg < DIM);
        if (!is_x && first) {
            v4bf z; z[0] = (__bf16)0.f; z[1] = (__bf16)0.f;
                    z[2] = (__bf16)0.f; z[3] = (__bf16)0.f;
            #pragma unroll
            for (int kk = 0; kk < 32; kk += 4) *(v4bf*)(dst + kk) = z;
        } else {
            const float* src = is_x ? (x_t    + (size_t)sb * DIM + kg)
                                    : (h_prev + (size_t)sb * HID + (kg - DIM));
            #pragma unroll
            for (int kk = 0; kk < 32; kk += 4) {
                float4 v = *(const float4*)(src + kk);
                v4bf p;
                p[0] = (__bf16)v.x; p[1] = (__bf16)v.y;
                p[2] = (__bf16)v.z; p[3] = (__bf16)v.w;
                *(v4bf*)(dst + kk) = p;
            }
        }
    };

    stage(0, 0);
    __syncthreads();

    for (int c = 0; c < NCHUNK; ++c) {
        const int buf = c & 1;
        if (c + 1 < NCHUNK) stage(c + 1, buf ^ 1);   // overlap with WMMAs below

        #pragma unroll
        for (int ks = 0; ks < 64; ks += 32) {
            // A fragment: lane holds row r, K = {hi*16..+15}
            const __bf16* abase = &smem[buf][r * 72 + ks + hi * 16];
            v8bf alo = *(const v8bf*)(abase);
            v8bf ahi = *(const v8bf*)(abase + 8);
            v16bf a;
            #pragma unroll
            for (int e = 0; e < 8; ++e) { a[e] = alo[e]; a[e + 8] = ahi[e]; }

            int kt = (c * 64 + ks) >> 5;
            const __bf16* wb =
                wpack + (((size_t)n0 * 64 + kt) * 32 + lane) * 16; // gate 0 base
            v16bf b0 = *(const v16bf*)(wb);
            v16bf b1 = *(const v16bf*)(wb +     gate_stride);
            v16bf b2 = *(const v16bf*)(wb + 2 * gate_stride);
            v16bf b3 = *(const v16bf*)(wb + 3 * gate_stride);

            accf = __builtin_amdgcn_wmma_f32_16x16x32_bf16(false, a, false, b0,
                                                           (short)0, accf, false, false);
            acci = __builtin_amdgcn_wmma_f32_16x16x32_bf16(false, a, false, b1,
                                                           (short)0, acci, false, false);
            accg = __builtin_amdgcn_wmma_f32_16x16x32_bf16(false, a, false, b2,
                                                           (short)0, accg, false, false);
            acco = __builtin_amdgcn_wmma_f32_16x16x32_bf16(false, a, false, b3,
                                                           (short)0, acco, false, false);
        }
        __syncthreads();
    }

    // Pointwise LSTM update: lane, vgpr j -> row mb + hi*8 + j, col n0*16 + r
    const int nh = n0 * 16 + r;
    const float bF = bF_[nh], bI = bI_[nh], bG = bG_[nh], bO = bO_[nh];
    #pragma unroll
    for (int j = 0; j < 8; ++j) {
        int bb = mb + hi * 8 + j;
        size_t off = (size_t)bb * HID + nh;
        float f = sigmoid_f(accf[j] + bF);
        float i = sigmoid_f(acci[j] + bI);
        float g = tanh_f   (accg[j] + bG);
        float o = sigmoid_f(acco[j] + bO);
        float cold;
        if (first) cold = 0.0f; else cold = c_state[off];
        float cn = f * cold + i * g;
        c_state[off] = cn;
        h_out[off]   = o * tanh_f(cn);
    }
}

// append (hT, cT) after outputs
__global__ __launch_bounds__(256) void finalize_states(
    const float* __restrict__ h_last, const float* __restrict__ c_state,
    float* __restrict__ tail)
{
    int i = blockIdx.x * 256 + threadIdx.x;
    if (i < BATCH * HID) {
        tail[i]               = h_last[i];
        tail[BATCH * HID + i] = c_state[i];
    }
}

extern "C" void kernel_launch(void* const* d_in, const int* in_sizes, int n_in,
                              void* d_out, int out_size, void* d_ws, size_t ws_size,
                              hipStream_t stream) {
    (void)in_sizes; (void)n_in; (void)out_size; (void)ws_size;
    const float* inputs = (const float*)d_in[0];
    const float* Wf = (const float*)d_in[1];
    const float* bF = (const float*)d_in[2];
    const float* Wi = (const float*)d_in[3];
    const float* bI = (const float*)d_in[4];
    const float* Wg = (const float*)d_in[5];
    const float* bG = (const float*)d_in[6];
    const float* Wo = (const float*)d_in[7];
    const float* bO = (const float*)d_in[8];

    float*  out     = (float*)d_out;
    __bf16* wpack   = (__bf16*)d_ws;
    float*  c_state = (float*)((char*)d_ws + (size_t)16 * 1024 * 1024);

    pack_weights<<<(PACK_ELEMS + 255) / 256, 256, 0, stream>>>(Wf, Wi, Wg, Wo, wpack);

    for (int t = 0; t < T_STEPS; ++t) {
        const float* x_t    = inputs + (size_t)t * BATCH * DIM;
        const float* h_prev = (t == 0) ? out : out + (size_t)(t - 1) * BATCH * HID;
        lstm_step<<<dim3(64, 4), 64, 0, stream>>>(
            x_t, h_prev, wpack, bF, bI, bG, bO,
            c_state, out + (size_t)t * BATCH * HID, (t == 0) ? 1 : 0);
    }

    finalize_states<<<(BATCH * HID + 255) / 256, 256, 0, stream>>>(
        out + (size_t)(T_STEPS - 1) * BATCH * HID, c_state,
        out + (size_t)T_STEPS * BATCH * HID);
}
